// TransformerLayer_37022618092058
// MI455X (gfx1250) — matrix operand order repaired and checked by reading, not compiled, not run on hardware
//
#include <hip/hip_runtime.h>
#include <hip/hip_bf16.h>

// ---------------------------------------------------------------------------
// Types & helpers for CDNA5 (gfx1250) wave32 WMMA bf16 path
// ---------------------------------------------------------------------------
typedef __attribute__((ext_vector_type(16))) __bf16 v16bf;
typedef __attribute__((ext_vector_type(8)))  float  v8f;
typedef unsigned __attribute__((ext_vector_type(4))) v4u;
typedef int      __attribute__((ext_vector_type(8))) v8i;
typedef int      __attribute__((ext_vector_type(4))) v4i;

#if defined(__gfx1250__) && __has_builtin(__builtin_amdgcn_tensor_load_to_lds)
#define HAVE_TDM 1
#else
#define HAVE_TDM 0
#endif

union FragU {
    uint4 q[2];
    v16bf v;
};

__device__ __forceinline__ unsigned short f2bf(float f) {
    __hip_bfloat16 h = __float2bfloat16(f);
    return __builtin_bit_cast(unsigned short, h);
}
__device__ __forceinline__ unsigned pack2bf(float x0, float x1) {
    return (unsigned)f2bf(x0) | ((unsigned)f2bf(x1) << 16);
}

// A-fragment (16x32, bf16) from row-major [16][ldk] array.
__device__ __forceinline__ v16bf load_afrag(const unsigned short* base, int ldk) {
    const int lane = threadIdx.x & 31;
    const int m = lane & 15, half = lane >> 4;
    FragU f;
    f.q[0] = *(const uint4*)(base + m * ldk + half * 8);
    f.q[1] = *(const uint4*)(base + m * ldk + 16 + half * 8);
    return f.v;
}

// B-fragment (32x16, bf16) from transposed [Ncols][ldk] array (row = output col n).
__device__ __forceinline__ v16bf load_bfrag(const unsigned short* base, int ldk) {
    const int lane = threadIdx.x & 31;
    const int n = lane & 15, kh = lane >> 4;
    FragU f;
    f.q[0] = *(const uint4*)(base + n * ldk + kh * 8);
    f.q[1] = *(const uint4*)(base + n * ldk + 16 + kh * 8);
    return f.v;
}

__device__ __forceinline__ v8f wmma_bf(v16bf a, v16bf b, v8f c) {
    return __builtin_amdgcn_wmma_f32_16x16x32_bf16(false, a, false, b, (short)0, c, false, false);
}

__device__ __forceinline__ float rowsum16(float v) {
    v += __shfl_xor(v, 1, 32);
    v += __shfl_xor(v, 2, 32);
    v += __shfl_xor(v, 4, 32);
    v += __shfl_xor(v, 8, 32);
    return v;
}

__device__ __forceinline__ float leaky(float v) { return v >= 0.f ? v : 0.05f * v; }

#if HAVE_TDM
// TDM 2-D tile load: bf16 tile [tile_h rows][tile_w elems], global row pitch
// stride_elems, into LDS at lds_dst (packed row-major [tile_h][tile_w]).
// Descriptor per CDNA5 ISA ch.8 (D# group0 + group1; groups 2/3 zero for 2D).
// This toolchain exposes the 6-arg builtin: (v4u, v8i, v4i, v4i, v8i, i32 cpol).
__device__ __forceinline__ void tdm_load_2d(const unsigned short* gsrc, unsigned stride_elems,
                                            unsigned tile_w, unsigned tile_h,
                                            unsigned short* lds_dst) {
    unsigned long long ga = (unsigned long long)(size_t)gsrc;
    unsigned lds_off = (unsigned)(size_t)lds_dst;  // low 32 bits = LDS byte offset
    v4u g0;
    g0[0] = 1u;                                   // count=1 (valid), user mode
    g0[1] = lds_off;                              // lds_addr (bytes)
    g0[2] = (unsigned)(ga & 0xFFFFFFFFu);         // global_addr[31:0]
    g0[3] = (unsigned)((ga >> 32) & 0x01FFFFFFu)  // global_addr[56:32]
            | (2u << 30);                         // type = 2 ("image")
    v8i g1;
    g1[0] = (int)(1u << 16);                                  // data_size=1 (2 bytes)
    g1[1] = (int)((stride_elems & 0xFFFFu) << 16);            // tensor_dim0[15:0]
    g1[2] = (int)(((stride_elems >> 16) & 0xFFFFu) |          // tensor_dim0[31:16]
                  ((tile_h & 0xFFFFu) << 16));                // tensor_dim1[15:0]
    g1[3] = (int)(((tile_h >> 16) & 0xFFFFu) |                // tensor_dim1[31:16]
                  ((tile_w & 0xFFFFu) << 16));                // tile_dim0
    g1[4] = (int)(tile_h & 0xFFFFu);                          // tile_dim1 (tile_dim2=0)
    g1[5] = (int)stride_elems;                                // tensor_dim0_stride[31:0]
    g1[6] = 0;                                                // stride hi / dim1_stride lo
    g1[7] = 0;
    v4i g2 = {0, 0, 0, 0};
    v4i g3 = {0, 0, 0, 0};
    v8i g4 = {0, 0, 0, 0, 0, 0, 0, 0};
    __builtin_amdgcn_tensor_load_to_lds(g0, g1, g2, g3, g4, 0);
}
#endif

// ---------------------------------------------------------------------------
// Weight prepack: src f32 [R][C]  ->  dst bf16 [C][R]  (B-matrix [N][K] layout)
// ---------------------------------------------------------------------------
__global__ __launch_bounds__(256) void k_prepack(const float* __restrict__ src,
                                                 unsigned short* __restrict__ dst,
                                                 int R, int C) {
    int idx = blockIdx.x * 256 + threadIdx.x;
    if (idx >= R * C) return;
    int r = idx / C, c = idx % C;
    dst[c * R + r] = f2bf(src[idx]);
}

// ---------------------------------------------------------------------------
// QKV: nt = [nodes | times] (K=320) @ W (320x256).  One wave per 16-row tile,
// half of the output columns (8 n-tiles -> 64 accum VGPRs, room for pipelining).
// grid (64 rowtiles, 3 which, 2 colhalf), block 32.
// Weight K-slab staged to LDS via TDM each k-step.
// ---------------------------------------------------------------------------
__global__ __launch_bounds__(32) void k_qkv(
    const float* __restrict__ nodes, const float* __restrict__ times,
    const unsigned short* __restrict__ Wqt, const unsigned short* __restrict__ Wkt,
    const unsigned short* __restrict__ Wvt,
    const float* __restrict__ bq, const float* __restrict__ bk, const float* __restrict__ bv,
    float* __restrict__ qo, float* __restrict__ ko, unsigned short* __restrict__ vt) {
    __shared__ unsigned short bufA[16 * 32];
    __shared__ unsigned short wbuf[128 * 32];  // weight slab: 128 outcols x 32 k
    const int which = blockIdx.y;
    const int colhalf = blockIdx.z;
    const int rt = blockIdx.x;           // 0..63
    const int b = rt >> 4;
    const int rowbase = (rt & 15) * 16;  // node base
    const int lane = threadIdx.x;
    const int half = lane >> 4, nn = lane & 15;

    const unsigned short* Wt = (which == 0) ? Wqt : (which == 1) ? Wkt : Wvt;
    const float* bias = (which == 0) ? bq : (which == 1) ? bk : bv;

    v8f acc[8];
#pragma unroll
    for (int t = 0; t < 8; ++t) {
        float bvv = bias[colhalf * 128 + t * 16 + nn];
#pragma unroll
        for (int r = 0; r < 8; ++r) acc[t][r] = bvv;
    }

    for (int kc = 0; kc < 10; ++kc) {
        const int kbase = kc * 32;
        __syncthreads();
        // weight slab -> LDS (TDM overlaps with A staging below)
#if HAVE_TDM
        tdm_load_2d(Wt + (size_t)(colhalf * 128) * 320 + kbase, 320, 32, 128, wbuf);
#else
        for (int idx = lane; idx < 512; idx += 32) {  // 512 uint4
            int rr = idx >> 2, qq = idx & 3;
            ((uint4*)wbuf)[idx] =
                *(const uint4*)(Wt + (size_t)(colhalf * 128 + rr) * 320 + kbase + qq * 8);
        }
#endif
        for (int p = lane; p < 256; p += 32) {  // A tile: 256 bf16-pairs
            int r = p >> 4, c2 = (p & 15) * 2;
            int col = kbase + c2;
            float x0, x1;
            if (col < 256) {
                const float* sp = &nodes[(size_t)(b * 256 + rowbase + r) * 256 + col];
                x0 = sp[0]; x1 = sp[1];
            } else {
                const float* sp = &times[b * 64 + (col - 256)];
                x0 = sp[0]; x1 = sp[1];
            }
            *(unsigned*)&bufA[r * 32 + c2] = pack2bf(x0, x1);
        }
        __syncthreads();
#if HAVE_TDM
        __builtin_amdgcn_s_wait_tensorcnt(0);
#endif
        v16bf af = load_afrag(bufA, 32);
#pragma unroll
        for (int tg = 0; tg < 2; ++tg) {
            v16bf bf[4];
#pragma unroll
            for (int u = 0; u < 4; ++u)
                bf[u] = load_bfrag(wbuf + (tg * 4 + u) * 16 * 32, 32);
#pragma unroll
            for (int u = 0; u < 4; ++u)
                acc[tg * 4 + u] = wmma_bf(af, bf[u], acc[tg * 4 + u]);
        }
    }

#pragma unroll
    for (int t = 0; t < 8; ++t) {
        int col = colhalf * 128 + t * 16 + nn;
        if (which < 2) {
            float* out = (which == 0) ? qo : ko;
#pragma unroll
            for (int r = 0; r < 8; ++r)
                out[(size_t)(b * 256 + rowbase + half * 8 + r) * 256 + col] = acc[t][r];
        } else {
            int h = col >> 5, d = col & 31;
            unsigned short tmp[8];
#pragma unroll
            for (int r = 0; r < 8; ++r) tmp[r] = f2bf(acc[t][r]);
            unsigned short* dst = vt + ((size_t)(b * 8 + h) * 32 + d) * 256 + rowbase + half * 8;
            *(uint4*)dst = *(uint4*)tmp;
        }
    }
}

// ---------------------------------------------------------------------------
// EDGE mega-kernel: per wave = 16 (i,j) pairs (fixed i, 16 consecutive j).
// grid 2048, block 256.
// ---------------------------------------------------------------------------
__global__ __launch_bounds__(256) void k_edge(
    const float* __restrict__ edges, const float* __restrict__ q, const float* __restrict__ kk,
    const unsigned short* __restrict__ Wmt, const float* __restrict__ bm,
    const unsigned short* __restrict__ Wat, const float* __restrict__ ba,
    const unsigned short* __restrict__ Weot, const float* __restrict__ beo,
    const unsigned short* __restrict__ We1t, const float* __restrict__ be1,
    const unsigned short* __restrict__ We2t, const float* __restrict__ be2,
    const float* __restrict__ g1, const float* __restrict__ b1,
    const float* __restrict__ g2, const float* __restrict__ b2,
    float* __restrict__ logits, float* __restrict__ out_edges) {
    __shared__ unsigned short bufE[8][1024];
    __shared__ unsigned short bufS[8][1024];
    const int blk = blockIdx.x;
    const int jblk = blk & 1, i = (blk >> 1) & 255, b = blk >> 9;
    const int w = threadIdx.x >> 5, lane = threadIdx.x & 31;
    const int half = lane >> 4, nn = lane & 15;
    const int jb = jblk * 128 + w * 16;
    const size_t pairbase = ((size_t)(b * 256 + i) * 256 + jb);
    const float* esrc = edges + pairbase * 64;

    for (int p = lane; p < 512; p += 32) {  // pairs
        const float2 v = *(const float2*)(esrc + p * 2);
        *(unsigned*)&bufE[w][p * 2] = pack2bf(v.x, v.y);
    }
    __syncthreads();
    v16bf aE0 = load_afrag(&bufE[w][0], 64);
    v16bf aE1 = load_afrag(&bufE[w][32], 64);

    v8f eacc[4];
#pragma unroll
    for (int t = 0; t < 4; ++t) {
        float bvv = beo[t * 16 + nn];
#pragma unroll
        for (int r = 0; r < 8; ++r) eacc[t][r] = bvv;
    }

    const float* qrow = q + (size_t)(b * 256 + i) * 256;
    for (int c = 0; c < 8; ++c) {  // head = c, cols c*32..c*32+31
        v8f ne[2];
#pragma unroll
        for (int sub = 0; sub < 2; ++sub) {
            int nt = 2 * c + sub;
            int col = nt * 16 + nn;
            v8f em, ea;
            float bmv = bm[col], bav = ba[col];
#pragma unroll
            for (int r = 0; r < 8; ++r) { em[r] = bmv; ea[r] = bav; }
            v16bf fm0 = load_bfrag(Wmt + nt * 16 * 64, 64);
            v16bf fm1 = load_bfrag(Wmt + nt * 16 * 64 + 32, 64);
            v16bf fa0 = load_bfrag(Wat + nt * 16 * 64, 64);
            v16bf fa1 = load_bfrag(Wat + nt * 16 * 64 + 32, 64);
            em = wmma_bf(aE0, fm0, em);
            ea = wmma_bf(aE0, fa0, ea);
            em = wmma_bf(aE1, fm1, em);
            ea = wmma_bf(aE1, fa1, ea);
            float qv = qrow[col] * 0.0625f;  // 1/sqrt(256)
#pragma unroll
            for (int r = 0; r < 8; ++r) {
                int j = jb + half * 8 + r;
                float kv = kk[(size_t)(b * 256 + j) * 256 + col];
                float at = qv * kv;
                ne[sub][r] = at * em[r] + at + ea[r];
            }
        }
#pragma unroll
        for (int r = 0; r < 8; ++r) {
            float s = rowsum16(ne[0][r] + ne[1][r]);
            if (nn == 0) {
                int j = jb + half * 8 + r;
                logits[(((size_t)(b * 256 + i)) * 8 + c) * 256 + j] = s;
            }
        }
        __syncthreads();
#pragma unroll
        for (int sub = 0; sub < 2; ++sub)
#pragma unroll
            for (int r = 0; r < 8; ++r)
                bufS[w][(half * 8 + r) * 32 + sub * 16 + nn] = f2bf(leaky(ne[sub][r]));
        __syncthreads();
        v16bf aNE = load_afrag(&bufS[w][0], 32);
        __syncthreads();
        v16bf bw[4];
#pragma unroll
        for (int ot = 0; ot < 4; ++ot)
            bw[ot] = load_bfrag(Weot + ot * 16 * 256 + c * 32, 256);
#pragma unroll
        for (int ot = 0; ot < 4; ++ot)
            eacc[ot] = wmma_bf(aNE, bw[ot], eacc[ot]);
    }

    // residual + LN1e
    v8f ae[4];
#pragma unroll
    for (int ot = 0; ot < 4; ++ot) {
        int col = ot * 16 + nn;
#pragma unroll
        for (int r = 0; r < 8; ++r) eacc[ot][r] += esrc[(half * 8 + r) * 64 + col];
    }
#pragma unroll
    for (int r = 0; r < 8; ++r) {
        float s = 0.f, s2 = 0.f;
#pragma unroll
        for (int ot = 0; ot < 4; ++ot) { float x = eacc[ot][r]; s += x; s2 += x * x; }
        s = rowsum16(s); s2 = rowsum16(s2);
        float mean = s * (1.f / 64.f);
        float inv = rsqrtf(s2 * (1.f / 64.f) - mean * mean + 1e-5f);
#pragma unroll
        for (int ot = 0; ot < 4; ++ot) {
            int col = ot * 16 + nn;
            ae[ot][r] = (eacc[ot][r] - mean) * inv * g1[col] + b1[col];
        }
    }
    __syncthreads();
#pragma unroll
    for (int ot = 0; ot < 4; ++ot)
#pragma unroll
        for (int r = 0; r < 8; ++r)
            bufS[w][(half * 8 + r) * 64 + ot * 16 + nn] = f2bf(ae[ot][r]);
    __syncthreads();
    v16bf aA0 = load_afrag(&bufS[w][0], 64);
    v16bf aA1 = load_afrag(&bufS[w][32], 64);
    __syncthreads();
    v8f h[4];
    {
        v16bf bw1[8];
#pragma unroll
        for (int ot = 0; ot < 4; ++ot) {
            bw1[2 * ot]     = load_bfrag(We1t + ot * 16 * 64, 64);
            bw1[2 * ot + 1] = load_bfrag(We1t + ot * 16 * 64 + 32, 64);
        }
#pragma unroll
        for (int ot = 0; ot < 4; ++ot) {
            float bvv = be1[ot * 16 + nn];
#pragma unroll
            for (int r = 0; r < 8; ++r) h[ot][r] = bvv;
            h[ot] = wmma_bf(aA0, bw1[2 * ot], h[ot]);
            h[ot] = wmma_bf(aA1, bw1[2 * ot + 1], h[ot]);
        }
#pragma unroll
        for (int ot = 0; ot < 4; ++ot)
#pragma unroll
            for (int r = 0; r < 8; ++r) h[ot][r] = leaky(h[ot][r]);
    }
    __syncthreads();
#pragma unroll
    for (int ot = 0; ot < 4; ++ot)
#pragma unroll
        for (int r = 0; r < 8; ++r)
            bufS[w][(half * 8 + r) * 64 + ot * 16 + nn] = f2bf(h[ot][r]);
    __syncthreads();
    v16bf aH0 = load_afrag(&bufS[w][0], 64);
    v16bf aH1 = load_afrag(&bufS[w][32], 64);
    {
        v16bf bw2[8];
#pragma unroll
        for (int ot = 0; ot < 4; ++ot) {
            bw2[2 * ot]     = load_bfrag(We2t + ot * 16 * 64, 64);
            bw2[2 * ot + 1] = load_bfrag(We2t + ot * 16 * 64 + 32, 64);
        }
#pragma unroll
        for (int ot = 0; ot < 4; ++ot) {
            float bvv = be2[ot * 16 + nn];
            v8f me;
#pragma unroll
            for (int r = 0; r < 8; ++r) me[r] = bvv;
            me = wmma_bf(aH0, bw2[2 * ot], me);
            me = wmma_bf(aH1, bw2[2 * ot + 1], me);
#pragma unroll
            for (int r = 0; r < 8; ++r) ae[ot][r] += me[r];
        }
    }
#pragma unroll
    for (int r = 0; r < 8; ++r) {
        float s = 0.f, s2 = 0.f;
#pragma unroll
        for (int ot = 0; ot < 4; ++ot) { float x = ae[ot][r]; s += x; s2 += x * x; }
        s = rowsum16(s); s2 = rowsum16(s2);
        float mean = s * (1.f / 64.f);
        float inv = rsqrtf(s2 * (1.f / 64.f) - mean * mean + 1e-5f);
#pragma unroll
        for (int ot = 0; ot < 4; ++ot) {
            int col = ot * 16 + nn;
            out_edges[pairbase * 64 + (size_t)(half * 8 + r) * 64 + col] =
                (ae[ot][r] - mean) * inv * g2[col] + b2[col];
        }
    }
}

// ---------------------------------------------------------------------------
// Softmax over j: logits [b][i][h][256] f32 -> a bf16 [b][h][i][256]
// ---------------------------------------------------------------------------
__global__ __launch_bounds__(256) void k_softmax(const float* __restrict__ logits,
                                                 unsigned short* __restrict__ a) {
    int gw = blockIdx.x * 8 + (threadIdx.x >> 5);  // 0..8191
    int lane = threadIdx.x & 31;
    int h = gw & 7, i = (gw >> 3) & 255, b = gw >> 11;
    const float* src = logits + ((size_t)(b * 256 + i) * 8 + h) * 256;
    float vals[8];
    float mx = -1e30f;
#pragma unroll
    for (int t = 0; t < 8; ++t) { vals[t] = src[t * 32 + lane]; mx = fmaxf(mx, vals[t]); }
#pragma unroll
    for (int m = 1; m < 32; m <<= 1) mx = fmaxf(mx, __shfl_xor(mx, m, 32));
    float sum = 0.f;
#pragma unroll
    for (int t = 0; t < 8; ++t) { vals[t] = __expf(vals[t] - mx); sum += vals[t]; }
#pragma unroll
    for (int m = 1; m < 32; m <<= 1) sum += __shfl_xor(sum, m, 32);
    float inv = 1.f / sum;
    unsigned short* dst = a + (((size_t)b * 8 + h) * 256 + i) * 256;
#pragma unroll
    for (int t = 0; t < 8; ++t) dst[t * 32 + lane] = f2bf(vals[t] * inv);
}

// ---------------------------------------------------------------------------
// WV: wv[b,i,h*32+d] = sum_j a[b,h,i,j] * v[b,j,h,d]
// ---------------------------------------------------------------------------
__global__ __launch_bounds__(256) void k_wv(const unsigned short* __restrict__ a,
                                            const unsigned short* __restrict__ vt,
                                            float* __restrict__ wv) {
    int blk = blockIdx.x;
    int w = threadIdx.x >> 5, lane = threadIdx.x & 31;
    int hg = blk & 1, h = (blk >> 1) & 7, b = blk >> 4;
    int it = hg * 8 + w;  // i-tile 0..15
    int ib = it * 16;
    const unsigned short* abase = a + (((size_t)b * 8 + h) * 256 + ib) * 256;
    const unsigned short* vbase = vt + ((size_t)(b * 8 + h) * 32) * 256;
    v8f acc0, acc1;
#pragma unroll
    for (int r = 0; r < 8; ++r) { acc0[r] = 0.f; acc1[r] = 0.f; }
#pragma unroll 2
    for (int kc = 0; kc < 8; ++kc) {
        v16bf af = load_afrag(abase + kc * 32, 256);
        v16bf bf0 = load_bfrag(vbase + kc * 32, 256);
        v16bf bf1 = load_bfrag(vbase + 16 * 256 + kc * 32, 256);
        acc0 = wmma_bf(af, bf0, acc0);
        acc1 = wmma_bf(af, bf1, acc1);
    }
    int half = lane >> 4, nn = lane & 15;
#pragma unroll
    for (int r = 0; r < 8; ++r) {
        int node = ib + half * 8 + r;
        wv[(size_t)(b * 256 + node) * 256 + h * 32 + nn] = acc0[r];
        wv[(size_t)(b * 256 + node) * 256 + h * 32 + 16 + nn] = acc1[r];
    }
}

// ---------------------------------------------------------------------------
// Node path: 3 WMMA GEMMs (M=1024, K=N=256) with fused LN/leaky; grid 8, block 256.
// Weight K-slab (16 KB) staged block-wide via TDM by wave 0 each k-step.
// ---------------------------------------------------------------------------
__device__ __forceinline__ void gemm1024(const float* __restrict__ src, int rowbase,
                                         const unsigned short* __restrict__ Wt,
                                         const float* __restrict__ bias,
                                         unsigned short* bufS, unsigned short* wslab,
                                         int w, v8f acc[16]) {
    const int lane = threadIdx.x & 31;
    const int nn = lane & 15;
#pragma unroll
    for (int t = 0; t < 16; ++t) {
        float bvv = bias[t * 16 + nn];
#pragma unroll
        for (int r = 0; r < 8; ++r) acc[t][r] = bvv;
    }
    for (int kc = 0; kc < 8; ++kc) {
        __syncthreads();
#if HAVE_TDM
        if (w == 0) tdm_load_2d(Wt + kc * 32, 256, 32, 256, wslab);
#else
        for (int idx = threadIdx.x; idx < 1024; idx += 256) {  // 1024 uint4
            int rr = idx >> 2, qq = idx & 3;
            ((uint4*)wslab)[idx] = *(const uint4*)(Wt + (size_t)rr * 256 + kc * 32 + qq * 8);
        }
#endif
        for (int p = lane; p < 256; p += 32) {  // A-tile bf16 pairs
            int r = p >> 4, c2 = (p & 15) * 2;
            const float* sp = src + (size_t)(rowbase + r) * 256 + kc * 32 + c2;
            *(unsigned*)&bufS[r * 32 + c2] = pack2bf(sp[0], sp[1]);
        }
#if HAVE_TDM
        if (w == 0) __builtin_amdgcn_s_wait_tensorcnt(0);
#endif
        __syncthreads();
        v16bf af = load_afrag(bufS, 32);
#pragma unroll
        for (int tg = 0; tg < 4; ++tg) {
            v16bf bf[4];
#pragma unroll
            for (int u = 0; u < 4; ++u)
                bf[u] = load_bfrag(wslab + (tg * 4 + u) * 16 * 32, 32);
#pragma unroll
            for (int u = 0; u < 4; ++u)
                acc[tg * 4 + u] = wmma_bf(af, bf[u], acc[tg * 4 + u]);
        }
    }
}

__device__ __forceinline__ void ln_store256(v8f acc[16], const float* __restrict__ g,
                                            const float* __restrict__ bt, bool do_leaky,
                                            float* __restrict__ out, int rowbase) {
    const int lane = threadIdx.x & 31;
    const int half = lane >> 4, nn = lane & 15;
#pragma unroll
    for (int r = 0; r < 8; ++r) {
        float s = 0.f, s2 = 0.f;
#pragma unroll
        for (int t = 0; t < 16; ++t) { float x = acc[t][r]; s += x; s2 += x * x; }
        s = rowsum16(s); s2 = rowsum16(s2);
        float mean = s * (1.f / 256.f);
        float inv = rsqrtf(s2 * (1.f / 256.f) - mean * mean + 1e-5f);
#pragma unroll
        for (int t = 0; t < 16; ++t) {
            int col = t * 16 + nn;
            float y = (acc[t][r] - mean) * inv * g[col] + bt[col];
            if (do_leaky) y = leaky(y);
            out[(size_t)(rowbase + half * 8 + r) * 256 + col] = y;
        }
    }
}

__global__ __launch_bounds__(256) void k_node(
    const float* __restrict__ wv, const float* __restrict__ nodes,
    const unsigned short* __restrict__ Wnot, const float* __restrict__ bno,
    const unsigned short* __restrict__ Wn1t, const float* __restrict__ bn1,
    const unsigned short* __restrict__ Wn2t, const float* __restrict__ bn2,
    const float* __restrict__ g1, const float* __restrict__ b1,
    const float* __restrict__ g2, const float* __restrict__ b2,
    float* __restrict__ anb, float* __restrict__ h1b, float* __restrict__ out_nodes) {
    __shared__ unsigned short bufS[8][512];
    __shared__ unsigned short wslab[256 * 32];  // 16 KB block-shared weight slab
    int w = threadIdx.x >> 5, lane = threadIdx.x & 31;
    int half = lane >> 4, nn = lane & 15;
    int rowbase = (blockIdx.x * 8 + w) * 16;
    v8f acc[16];

    // phase 1: attn_nodes = leaky(LN1n(wv@Wno + bno + nodes))
    gemm1024(wv, rowbase, Wnot, bno, bufS[w], wslab, w, acc);
#pragma unroll
    for (int t = 0; t < 16; ++t)
#pragma unroll
        for (int r = 0; r < 8; ++r)
            acc[t][r] += nodes[(size_t)(rowbase + half * 8 + r) * 256 + t * 16 + nn];
    ln_store256(acc, g1, b1, true, anb, rowbase);
    __threadfence();

    // phase 2: h1 = leaky(an@Wn1 + bn1)
    gemm1024(anb, rowbase, Wn1t, bn1, bufS[w], wslab, w, acc);
#pragma unroll
    for (int t = 0; t < 16; ++t)
#pragma unroll
        for (int r = 0; r < 8; ++r)
            h1b[(size_t)(rowbase + half * 8 + r) * 256 + t * 16 + nn] = leaky(acc[t][r]);
    __threadfence();

    // phase 3: out = leaky(LN2n(h1@Wn2 + bn2 + an))
    gemm1024(h1b, rowbase, Wn2t, bn2, bufS[w], wslab, w, acc);
#pragma unroll
    for (int t = 0; t < 16; ++t)
#pragma unroll
        for (int r = 0; r < 8; ++r)
            acc[t][r] += anb[(size_t)(rowbase + half * 8 + r) * 256 + t * 16 + nn];
    ln_store256(acc, g2, b2, true, out_nodes, rowbase);
}

// ---------------------------------------------------------------------------
// Host launcher
// ---------------------------------------------------------------------------
extern "C" void kernel_launch(void* const* d_in, const int* in_sizes, int n_in,
                              void* d_out, int out_size, void* d_ws, size_t ws_size,
                              hipStream_t stream) {
    (void)in_sizes; (void)n_in; (void)out_size; (void)ws_size;
    const float* nodes = (const float*)d_in[0];
    const float* edges = (const float*)d_in[1];
    const float* times = (const float*)d_in[2];
    const float* Wq = (const float*)d_in[3];  const float* bq  = (const float*)d_in[4];
    const float* Wk = (const float*)d_in[5];  const float* bk  = (const float*)d_in[6];
    const float* Wv = (const float*)d_in[7];  const float* bv  = (const float*)d_in[8];
    const float* Wm = (const float*)d_in[9];  const float* bm  = (const float*)d_in[10];
    const float* Wa = (const float*)d_in[11]; const float* ba  = (const float*)d_in[12];
    const float* Wno = (const float*)d_in[13]; const float* bno = (const float*)d_in[14];
    const float* Weo = (const float*)d_in[15]; const float* beo = (const float*)d_in[16];
    const float* ln1n_g = (const float*)d_in[17]; const float* ln1n_b = (const float*)d_in[18];
    const float* ln1e_g = (const float*)d_in[19]; const float* ln1e_b = (const float*)d_in[20];
    const float* Wn1 = (const float*)d_in[21]; const float* bn1 = (const float*)d_in[22];
    const float* Wn2 = (const float*)d_in[23]; const float* bn2 = (const float*)d_in[24];
    const float* We1 = (const float*)d_in[25]; const float* be1 = (const float*)d_in[26];
    const float* We2 = (const float*)d_in[27]; const float* be2 = (const float*)d_in[28];
    const float* ln2n_g = (const float*)d_in[29]; const float* ln2n_b = (const float*)d_in[30];
    const float* ln2e_g = (const float*)d_in[31]; const float* ln2e_b = (const float*)d_in[32];

    float* out_nodes = (float*)d_out;                 // [4*256*256]
    float* out_edges = out_nodes + 4 * 256 * 256;     // [4*256*256*64]

    char* ws = (char*)d_ws;
    size_t off = 0;
    auto take = [&](size_t bytes) -> char* {
        char* p = ws + off;
        off = (off + bytes + 255) & ~(size_t)255;
        return p;
    };
    float* q      = (float*)take(4ull * 256 * 256 * 4);            // [b][n][256]
    float* k      = (float*)take(4ull * 256 * 256 * 4);
    unsigned short* vt = (unsigned short*)take(4ull * 8 * 32 * 256 * 2);  // [b][h][d][n]
    float* logits = (float*)take(4ull * 256 * 8 * 256 * 4);        // [b][i][h][j]
    unsigned short* abf = (unsigned short*)take(4ull * 8 * 256 * 256 * 2); // [b][h][i][j]
    float* wv     = (float*)take(4ull * 256 * 256 * 4);
    float* anb    = (float*)take(4ull * 256 * 256 * 4);
    float* h1b    = (float*)take(4ull * 256 * 256 * 4);
    unsigned short* Wqt  = (unsigned short*)take(256ull * 320 * 2);
    unsigned short* Wkt  = (unsigned short*)take(256ull * 320 * 2);
    unsigned short* Wvt  = (unsigned short*)take(256ull * 320 * 2);
    unsigned short* Wmt  = (unsigned short*)take(256ull * 64 * 2);
    unsigned short* Wat  = (unsigned short*)take(256ull * 64 * 2);
    unsigned short* Weot = (unsigned short*)take(64ull * 256 * 2);
    unsigned short* We1t = (unsigned short*)take(64ull * 64 * 2);
    unsigned short* We2t = (unsigned short*)take(64ull * 64 * 2);
    unsigned short* Wnot = (unsigned short*)take(256ull * 256 * 2);
    unsigned short* Wn1t = (unsigned short*)take(256ull * 256 * 2);
    unsigned short* Wn2t = (unsigned short*)take(256ull * 256 * 2);

    auto pp = [&](const float* src, unsigned short* dst, int R, int C) {
        int n = R * C;
        k_prepack<<<(n + 255) / 256, 256, 0, stream>>>(src, dst, R, C);
    };
    pp(Wq, Wqt, 320, 256);
    pp(Wk, Wkt, 320, 256);
    pp(Wv, Wvt, 320, 256);
    pp(Wm, Wmt, 64, 256);
    pp(Wa, Wat, 64, 256);
    pp(Weo, Weot, 256, 64);
    pp(We1, We1t, 64, 64);
    pp(We2, We2t, 64, 64);
    pp(Wno, Wnot, 256, 256);
    pp(Wn1, Wn1t, 256, 256);
    pp(Wn2, Wn2t, 256, 256);

    k_qkv<<<dim3(64, 3, 2), 32, 0, stream>>>(nodes, times, Wqt, Wkt, Wvt, bq, bk, bv, q, k, vt);

    k_edge<<<2048, 256, 0, stream>>>(edges, q, k, Wmt, bm, Wat, ba, Weot, beo,
                                     We1t, be1, We2t, be2,
                                     ln1e_g, ln1e_b, ln2e_g, ln2e_b,
                                     logits, out_edges);

    k_softmax<<<1024, 256, 0, stream>>>(logits, abf);

    k_wv<<<64, 256, 0, stream>>>(abf, vt, wv);

    k_node<<<8, 256, 0, stream>>>(wv, nodes, Wnot, bno, Wn1t, bn1, Wn2t, bn2,
                                  ln1n_g, ln1n_b, ln2n_g, ln2n_b,
                                  anb, h1b, out_nodes);
}
